// Attention_82154134438516
// MI455X (gfx1250) — matrix operand order repaired and checked by reading, not compiled, not run on hardware
//
#include <hip/hip_runtime.h>
#include <hip/hip_bf16.h>
#include <math.h>

// CDNA5 / gfx1250, wave32. Fused attention producing out (B,H,S,D) + p_attn (B,H,S,S).
// B=4, H=16, S=2048, D=64. fp32 in/out, bf16 WMMA internally.
// Scores computed TRANSPOSED (S^T tiles) so each lane owns 8 consecutive k of one q row:
// contiguous mask/sig loads, contiguous p_attn NT stores, scalar per-lane softmax stats.

typedef __attribute__((ext_vector_type(16))) __bf16 v16bf;
typedef __attribute__((ext_vector_type(8)))  float  v8f;
typedef __attribute__((ext_vector_type(4)))  float  v4f;   // native vector for NT stores

#define S_LEN 2048
#define D_DIM 64
#define H_NUM 16
#define B_NUM 4
#define KC    64          // keys per staged chunk
#define QROWS 128         // q rows per workgroup (8 waves * 16)
#define STR   72          // padded LDS row stride (bf16 elems): 144B -> conflict-free frags
#define SCALE 0.125f      // 1/sqrt(64), folded into Q at staging (exact pow2)

// ---- float -> bf16 (RNE, bit-level) ----
static __device__ __forceinline__ __bf16 f2bf(float x) {
  union { float f; unsigned u; } un; un.f = x;
  unsigned r = un.u + 0x7FFFu + ((un.u >> 16) & 1u);
  unsigned short h = (unsigned short)(r >> 16);
  __bf16 o; __builtin_memcpy(&o, &h, 2);
  return o;
}
static __device__ __forceinline__ unsigned pack2(float a, float b) {
  union { float f; unsigned u; } ua, ub; ua.f = a; ub.f = b;
  unsigned ra = (ua.u + 0x7FFFu + ((ua.u >> 16) & 1u)) >> 16;
  unsigned rb = (ub.u + 0x7FFFu + ((ub.u >> 16) & 1u)) & 0xFFFF0000u;
  return (ra & 0xFFFFu) | rb;
}

// A operand (M x 32): row-major [M][K] tile. ISA 16-bit A layout:
//  lanes 0-15: row=lane, v<4 -> K=2v..; v>=4 -> K=16+..; lanes 16-31: +8 K offset.
static __device__ __forceinline__ v16bf load_fragA(const __bf16* tile, int lane, int cOff) {
  const int r  = lane & 15;
  const int hi = (lane >> 4) & 1;
  const __bf16* row = tile + r * STR;
  v16bf f;
#pragma unroll
  for (int v = 0; v < 8; ++v) {
    int k = ((v < 4) ? (2 * v) : (16 + 2 * (v - 4))) + (hi ? 8 : 0) + cOff;
    f[2 * v]     = row[k];
    f[2 * v + 1] = row[k + 1];
  }
  return f;
}
// B operand (32 x N): tile stored TRANSPOSED row-major [N][K]. ISA B layout:
//  lanes 0-15 hold K=0..7,16..23; lanes 16-31 hold K=8..15,24..31 (matches A).
static __device__ __forceinline__ v16bf load_fragB(const __bf16* tileT, int lane, int cOff) {
  const int nn = lane & 15;
  const int hi = (lane >> 4) & 1;
  const __bf16* row = tileT + nn * STR + hi * 8 + cOff;
  v16bf f;
#pragma unroll
  for (int v = 0; v < 16; ++v) f[v] = row[(v < 8) ? v : (v + 8)];
  return f;
}

static __device__ __forceinline__ v8f wmma_bf16(v16bf a, v16bf b, v8f c) {
  return __builtin_amdgcn_wmma_f32_16x16x32_bf16(false, a, false, b,
                                                 (short)0, c, false, false);
}

// Stage KC x D chunk: sKT = bf16(K+T) row-major [k][d]; sVT = bf16(V) transposed [d][k].
static __device__ __forceinline__ void stage_chunk(const float* gK, const float* gT,
                                                   const float* gV, int kc0,
                                                   __bf16* sKT, __bf16* sVT,
                                                   bool loadV, int tid) {
#pragma unroll
  for (int it = 0; it < 4; ++it) {                 // 64*64 / 256 thr = 4x float4
    int e   = (tid + it * 256) * 4;
    int row = e >> 6;
    int col = e & 63;
    const float4 k4 = *(const float4*)(gK + (size_t)(kc0 + row) * D_DIM + col);
    const float4 t4 = *(const float4*)(gT + (size_t)(kc0 + row) * D_DIM + col);
    uint2 pk;
    pk.x = pack2(k4.x + t4.x, k4.y + t4.y);
    pk.y = pack2(k4.z + t4.z, k4.w + t4.w);
    *(uint2*)&sKT[row * STR + col] = pk;
    if (loadV) {
      const float4 v4 = *(const float4*)(gV + (size_t)(kc0 + row) * D_DIM + col);
      sVT[(col + 0) * STR + row] = f2bf(v4.x);
      sVT[(col + 1) * STR + row] = f2bf(v4.y);
      sVT[(col + 2) * STR + row] = f2bf(v4.z);
      sVT[(col + 3) * STR + row] = f2bf(v4.w);
    }
  }
}

template <bool SIGPRE>
__launch_bounds__(256)
__global__ void attn_fused_kernel(const float* __restrict__ Q,
                                  const float* __restrict__ K,
                                  const float* __restrict__ V,
                                  const float* __restrict__ T,
                                  const unsigned char* __restrict__ amask,
                                  const float* __restrict__ sigl,   // sigmoid table or raw logits
                                  float* __restrict__ out,
                                  float* __restrict__ pattn) {
  __shared__ __align__(16) __bf16 sQ[QROWS * STR];    // 18 KB  [q][d], *SCALE folded
  __shared__ __align__(16) __bf16 sKT[KC * STR];      //  9 KB  [k][d], K+T
  __shared__ __align__(16) __bf16 sVT[D_DIM * STR];   //  9 KB  [d][k]
  __shared__ __align__(16) __bf16 sP[8][16 * STR];    // 18 KB  per-wave [q][k]

  const int tid  = threadIdx.x;
  const int w    = tid >> 5;
  const int lane = tid & 31;
  const int n    = lane & 15;          // q row (scores) / d col (PV)
  const int hi   = (lane >> 4) & 1;

  const int bh   = blockIdx.x >> 4;              // b*H + h
  const int qblk = blockIdx.x & 15;
  const int b    = bh >> 4;                      // H = 16
  const int q0   = qblk * QROWS + w * 16;

  const size_t qkvBase = (size_t)bh * S_LEN * D_DIM;
  const float* gQ = Q + qkvBase;
  const float* gK = K + qkvBase;
  const float* gT = T + qkvBase;
  const float* gV = V + qkvBase;
  const unsigned char* gM = amask + (size_t)b * S_LEN * S_LEN;

  // ---- stage Q block (128x64) once, scale folded ----
#pragma unroll
  for (int it = 0; it < 8; ++it) {
    int e   = (tid + it * 256) * 4;
    int row = e >> 6;
    int col = e & 63;
    const float4 q4 = *(const float4*)(gQ + (size_t)(qblk * QROWS + row) * D_DIM + col);
    uint2 pk;
    pk.x = pack2(q4.x * SCALE, q4.y * SCALE);
    pk.y = pack2(q4.z * SCALE, q4.w * SCALE);
    *(uint2*)&sQ[row * STR + col] = pk;
  }
  __syncthreads();

  // Q used as B operand (tile is [q][d] row-major == [N][K] transposed form)
  const v16bf bQ0 = load_fragB(sQ + w * 16 * STR, lane, 0);
  const v16bf bQ1 = load_fragB(sQ + w * 16 * STR, lane, 32);

  const size_t mrowBase = (size_t)(q0 + n) * S_LEN;   // this lane's q row in mask/sig/pattn

  // ================= PASS 1: per-lane online row max / sum ==================
  float ml = -INFINITY, lsum = 0.0f;
  for (int kc0 = 0; kc0 < S_LEN; kc0 += KC) {
    __syncthreads();
    stage_chunk(gK, gT, gV, kc0, sKT, sVT, false, tid);
    __syncthreads();
#pragma unroll
    for (int j = 0; j < 4; ++j) {
      v8f acc = {};
      acc = wmma_bf16(load_fragA(sKT + (j * 16) * STR, lane, 0),  bQ0, acc);
      acc = wmma_bf16(load_fragA(sKT + (j * 16) * STR, lane, 32), bQ1, acc);
      const int kb = kc0 + j * 16 + hi * 8;            // 8 consecutive k for this lane
      const uint2 mb = *(const uint2*)(gM + mrowBase + kb);
      float sv[8];
#pragma unroll
      for (int r = 0; r < 8; ++r) {
        unsigned bit = (((r < 4) ? mb.x : mb.y) >> (8 * (r & 3))) & 1u;
        sv[r] = bit ? acc[r] : -1e9f;
      }
      float m8 = fmaxf(fmaxf(fmaxf(sv[0], sv[1]), fmaxf(sv[2], sv[3])),
                       fmaxf(fmaxf(sv[4], sv[5]), fmaxf(sv[6], sv[7])));
      float nm = fmaxf(ml, m8);
      float es = 0.0f;
#pragma unroll
      for (int r = 0; r < 8; ++r) es += __expf(sv[r] - nm);
      lsum = lsum * __expf(ml - nm) + es;
      ml = nm;
    }
  }
  // merge the two k-halves of this q row (lanes n and n+16), fold 1/L into the max
  {
    const float om = __shfl_xor(ml, 16, 32);
    const float os = __shfl_xor(lsum, 16, 32);
    const float M  = fmaxf(ml, om);
    const float L  = lsum * __expf(ml - M) + os * __expf(om - M);
    ml = M + __logf(L);                                // ml := Mlog
  }

  // ================= PASS 2: p = exp(s - Mlog) * sig; write P; P@V =========
  v8f accO[4] = {};
  float* pRow = pattn + ((size_t)bh * S_LEN) * S_LEN + mrowBase;
  for (int kc0 = 0; kc0 < S_LEN; kc0 += KC) {
    __syncthreads();
    stage_chunk(gK, gT, gV, kc0, sKT, sVT, true, tid);
    __syncthreads();
#pragma unroll
    for (int j = 0; j < 4; ++j) {
      v8f acc = {};
      acc = wmma_bf16(load_fragA(sKT + (j * 16) * STR, lane, 0),  bQ0, acc);
      acc = wmma_bf16(load_fragA(sKT + (j * 16) * STR, lane, 32), bQ1, acc);
      const int kb = kc0 + j * 16 + hi * 8;
      const uint2 mb = *(const uint2*)(gM + mrowBase + kb);
      float4 s0, s1;
      if (SIGPRE) {
        s0 = *(const float4*)(sigl + mrowBase + kb);
        s1 = *(const float4*)(sigl + mrowBase + kb + 4);
      } else {
        const float4 l0 = *(const float4*)(sigl + mrowBase + kb);
        const float4 l1 = *(const float4*)(sigl + mrowBase + kb + 4);
        s0.x = 1.0f / (1.0f + __expf(-l0.x)); s0.y = 1.0f / (1.0f + __expf(-l0.y));
        s0.z = 1.0f / (1.0f + __expf(-l0.z)); s0.w = 1.0f / (1.0f + __expf(-l0.w));
        s1.x = 1.0f / (1.0f + __expf(-l1.x)); s1.y = 1.0f / (1.0f + __expf(-l1.y));
        s1.z = 1.0f / (1.0f + __expf(-l1.z)); s1.w = 1.0f / (1.0f + __expf(-l1.w));
      }
      float p[8];
#pragma unroll
      for (int r = 0; r < 8; ++r) {
        unsigned bit = (((r < 4) ? mb.x : mb.y) >> (8 * (r & 3))) & 1u;
        float s = bit ? acc[r] : -1e9f;
        p[r] = __expf(s - ml);                          // ml == Mlog
      }
      p[0] *= s0.x; p[1] *= s0.y; p[2] *= s0.z; p[3] *= s0.w;
      p[4] *= s1.x; p[5] *= s1.y; p[6] *= s1.z; p[7] *= s1.w;
      // 1 GB write-once stream: nontemporal, two b128 per lane per subtile
      v4f o0; o0.x = p[0]; o0.y = p[1]; o0.z = p[2]; o0.w = p[3];
      v4f o1; o1.x = p[4]; o1.y = p[5]; o1.z = p[6]; o1.w = p[7];
      __builtin_nontemporal_store(o0, (v4f*)(pRow + kb));
      __builtin_nontemporal_store(o1, (v4f*)(pRow + kb + 4));
      // stage bf16 P row-major [q][k]; 8 consecutive b16 stores -> ds_store_b128
      __bf16* pd = &sP[w][n * STR + j * 16 + hi * 8];
#pragma unroll
      for (int r = 0; r < 8; ++r) pd[r] = f2bf(p[r]);
    }
    // out tile += P(16xKC) @ V(KCx64):  A = sP [q][k], B = sVT [d][k]
#pragma unroll
    for (int c = 0; c < 2; ++c) {
      const v16bf aP = load_fragA(sP[w], lane, c * 32);
#pragma unroll
      for (int j2 = 0; j2 < 4; ++j2) {
        const v16bf bV = load_fragB(sVT + (j2 * 16) * STR, lane, c * 32);
        accO[j2] = wmma_bf16(aP, bV, accO[j2]);
      }
    }
  }

  // ---- write out tile (16 x 64); rows m = r + hi*8, cols d = j2*16 + n ----
#pragma unroll
  for (int j2 = 0; j2 < 4; ++j2) {
#pragma unroll
    for (int r = 0; r < 8; ++r) {
      out[((size_t)bh * S_LEN + (q0 + r + hi * 8)) * D_DIM + j2 * 16 + n] = accO[j2][r];
    }
  }
}

// Precompute sigmoid(mask_logits) into workspace: 16 MB, L2-resident, reused 64x.
__launch_bounds__(256)
__global__ void sigmoid_kernel(const float* __restrict__ logits, float* __restrict__ sig) {
  const size_t i = ((size_t)blockIdx.x * 256 + threadIdx.x) * 4;
  const float4 l = *(const float4*)(logits + i);
  float4 o;
  o.x = 1.0f / (1.0f + __expf(-l.x));
  o.y = 1.0f / (1.0f + __expf(-l.y));
  o.z = 1.0f / (1.0f + __expf(-l.z));
  o.w = 1.0f / (1.0f + __expf(-l.w));
  *(float4*)(sig + i) = o;
}

extern "C" void kernel_launch(void* const* d_in, const int* in_sizes, int n_in,
                              void* d_out, int out_size, void* d_ws, size_t ws_size,
                              hipStream_t stream) {
  const float*         Q      = (const float*)d_in[0];
  const float*         K      = (const float*)d_in[1];
  const float*         V      = (const float*)d_in[2];
  const float*         T      = (const float*)d_in[3];
  const unsigned char* amask  = (const unsigned char*)d_in[4]; // bool (B,1,S,S)
  const float*         logits = (const float*)d_in[5];         // (S,S)

  float* out   = (float*)d_out;                                // (B,H,S,D)
  float* pattn = out + (size_t)B_NUM * H_NUM * S_LEN * D_DIM;  // (B,H,S,S)

  const int grid = B_NUM * H_NUM * (S_LEN / QROWS);            // 1024 WGs x 256 thr
  const size_t sigBytes = (size_t)S_LEN * S_LEN * sizeof(float);

  if (ws_size >= sigBytes) {
    float* sig = (float*)d_ws;
    sigmoid_kernel<<<dim3((S_LEN * S_LEN) / (256 * 4)), dim3(256), 0, stream>>>(logits, sig);
    attn_fused_kernel<true><<<dim3(grid), dim3(256), 0, stream>>>(
        Q, K, V, T, amask, sig, out, pattn);
  } else {
    attn_fused_kernel<false><<<dim3(grid), dim3(256), 0, stream>>>(
        Q, K, V, T, amask, logits, out, pattn);
  }
}
